// BiLSTMModel_45629732552714
// MI455X (gfx1250) — matrix operand order repaired and checked by reading, not compiled, not run on hardware
//
#include <hip/hip_runtime.h>
#include <hip/hip_bf16.h>

// ---------------------------------------------------------------------------
// Types for CDNA5 WMMA (wave32): D(f32 16x16) = A(bf16 16x32) * B(bf16 32x16) + C
// ---------------------------------------------------------------------------
typedef __bf16 bf16_t;
typedef __attribute__((ext_vector_type(16))) __bf16 v16bf;
typedef __attribute__((ext_vector_type(8)))  __bf16 v8bf;
typedef __attribute__((ext_vector_type(8)))  float  v8f;
typedef __attribute__((ext_vector_type(4)))  unsigned int u32x4;
typedef __attribute__((ext_vector_type(8)))  int          i32x8;
typedef __attribute__((ext_vector_type(4)))  int          i32x4;

__device__ __forceinline__ v8f wmma_bf16(v16bf a, v16bf b, v8f c) {
  // (neg_a, A, neg_b, B, c_mod, C, reuse_a, reuse_b)
  return __builtin_amdgcn_wmma_f32_16x16x32_bf16(false, a, false, b, (short)0, c,
                                                 false, false);
}

// A fragment: A is row-major [M,K] bf16. Layout used here (contiguous k, same as B) for 16-bit A 16x32:
//  lanes 0-15 : row m0+lane, K = k0..k0+15
//  lanes 16-31: row m0+(lane-16), K = k0+16..k0+31
__device__ __forceinline__ v16bf load_a_frag(const bf16_t* A, int lda, int m0,
                                             int k0, int lane) {
  int m  = m0 + (lane & 15);
  int kc = k0 + ((lane >> 4) << 4);
  const bf16_t* p = A + (size_t)m * lda + kc;
  v8bf lo = *(const v8bf*)(p);        // 16B
  v8bf hi = *(const v8bf*)(p + 8);    // 16B
  v16bf a;
#pragma unroll
  for (int i = 0; i < 8; ++i) { a[i] = lo[i]; a[i + 8] = hi[i]; }
  return a;
}

// B fragment: BT is row-major [N,K] bf16 (i.e. W^T). Layout used here (contiguous k, same as A) for 32x16 B:
//  lanes 0-15 : col n0+lane, K = k0..k0+15 ;  lanes 16-31: col, K = k0+16..k0+31
__device__ __forceinline__ v16bf load_b_frag(const bf16_t* BT, int ldb, int n0,
                                             int k0, int lane) {
  int n = n0 + (lane & 15);
  int k = k0 + ((lane >> 4) << 4);
  return *(const v16bf*)(BT + (size_t)n * ldb + k);   // 32B contiguous
}

__device__ __forceinline__ float sigmoidf_(float x) {
  return 1.0f / (1.0f + __expf(-x));
}

// ---------------------------------------------------------------------------
// TDM: DMA a [64 x KC=128] bf16 tile of BT (row-major [N,K]) into LDS.
// D# per CDNA5 ISA ch.8: group0 = {count/flags, lds_addr, global_addr, type=2},
// group1 = {data_size=2B, tensor_dim0=K, tensor_dim1=N, tile 128x64, stride=K},
// groups 2/3 = zero (2D tile). clang-23 6-arg builtin form (extra i32x8 + cpol).
// ---------------------------------------------------------------------------
__device__ __forceinline__ void tdm_load_b_tile(const bf16_t* lds_dst,
                                                const bf16_t* BT,
                                                int N, int K, int n0, int k0) {
  unsigned long long ga = (unsigned long long)(uintptr_t)BT
                        + ((unsigned long long)n0 * (unsigned)K + (unsigned)k0) * 2ull;
  unsigned lds_off = (unsigned)(uintptr_t)lds_dst;   // LDS aperture: addr[31:0]

  u32x4 g0;
  g0[0] = 1u;                                         // count=1, user mode
  g0[1] = lds_off;                                    // lds_addr (bytes)
  g0[2] = (unsigned)ga;                               // global_addr[31:0]
  g0[3] = (unsigned)((ga >> 32) & 0x1ffffffull)       // global_addr[56:32]
        | (2u << 30);                                 // type = 2 ("image")

  i32x8 g1;
  g1[0] = (int)(1u << 16);                            // data_size=1 (2B), wg_mask=0
  g1[1] = (int)(((unsigned)K & 0xffffu) << 16);       // tensor_dim0[15:0] @ bits63:48
  g1[2] = (int)((((unsigned)N & 0xffffu) << 16)       // tensor_dim1[15:0] @ bits95:80
        | (((unsigned)K >> 16) & 0xffffu));           // tensor_dim0[31:16]
  g1[3] = (int)((128u << 16)                          // tile_dim0 = 128 (K elems)
        | (((unsigned)N >> 16) & 0xffffu));           // tensor_dim1[31:16]
  g1[4] = 64;                                         // tile_dim1 = 64 rows, tile_dim2=0
  g1[5] = K;                                          // tensor_dim0_stride[31:0]
  g1[6] = 0;                                          // stride[47:32]=0, dim1_stride lo=0
  g1[7] = 0;

  i32x4 g2 = {0, 0, 0, 0};
  i32x4 g3 = {0, 0, 0, 0};
  i32x8 g4 = {0, 0, 0, 0, 0, 0, 0, 0};
  __builtin_amdgcn_tensor_load_to_lds(g0, g1, g2, g3, g4, 0);
}

// ---------------------------------------------------------------------------
// Transpose + convert: src [rows,cols] f32 -> dst [cols,rows] bf16
// ---------------------------------------------------------------------------
__global__ __launch_bounds__(256) void k_transpose_to_bf16(const float* __restrict__ src,
                                                           bf16_t* __restrict__ dst,
                                                           int rows, int cols) {
  __shared__ float tile[32][33];
  int c0 = blockIdx.x * 32, r0 = blockIdx.y * 32;
  int tx = threadIdx.x & 31, ty = threadIdx.x >> 5;
#pragma unroll
  for (int i = 0; i < 32; i += 8)
    tile[ty + i][tx] = src[(size_t)(r0 + ty + i) * cols + c0 + tx];
  __syncthreads();
#pragma unroll
  for (int i = 0; i < 32; i += 8)
    dst[(size_t)(c0 + ty + i) * rows + r0 + tx] = (bf16_t)tile[tx][ty + i];
}

// ---------------------------------------------------------------------------
// Embedding gather + f32->bf16
// ---------------------------------------------------------------------------
__global__ void k_embed_bf16(const int* __restrict__ idx, const float* __restrict__ emb,
                             bf16_t* __restrict__ dst, int E) {
  int row = blockIdx.x;
  int v = idx[row];
  const float* s = emb + (size_t)v * E;
  bf16_t* d = dst + (size_t)row * E;
  for (int c = threadIdx.x; c < E; c += blockDim.x) d[c] = (bf16_t)s[c];
}

// ---------------------------------------------------------------------------
// GEMM: C[M,N] f32 = A[M,K] bf16 * BT[N,K]^T bf16 + bias[N]
// Block = 8 waves, wave computes one 16x16 tile, block tile 32(M) x 64(N).
// B is staged through LDS by the Tensor Data Mover, double-buffered in
// KC=128 chunks (16 KB per buffer), wave 0 drives the DMA + TENSORcnt.
// grid = (N/64, M/32); K % 128 == 0 for all our shapes.
// ---------------------------------------------------------------------------
__global__ __launch_bounds__(256) void k_gemm_bf16_tdm(const bf16_t* __restrict__ A,
                                                       const bf16_t* __restrict__ BT,
                                                       const float* __restrict__ bias,
                                                       float* __restrict__ C,
                                                       int M, int N, int K) {
  const int KC = 128;
  __shared__ bf16_t bsh[2][64 * KC];   // 2 x 16 KB
  int lane = threadIdx.x & 31;
  int wave = threadIdx.x >> 5;
  int m0  = blockIdx.y * 32 + (wave >> 2) * 16;
  int n0b = blockIdx.x * 64;           // block's N origin (global)
  int n0l = (wave & 3) * 16;           // wave's N origin inside LDS tile

  v8f acc;
#pragma unroll
  for (int i = 0; i < 8; ++i) acc[i] = 0.0f;

  int nchunks = K / KC;
  if (wave == 0) tdm_load_b_tile(bsh[0], BT, N, K, n0b, 0);

  for (int c = 0; c < nchunks; ++c) {
    if (wave == 0) {
      if (c + 1 < nchunks) {
        tdm_load_b_tile(bsh[(c + 1) & 1], BT, N, K, n0b, (c + 1) * KC);
        __builtin_amdgcn_s_wait_tensorcnt(1);  // in-order: chunk c has landed
      } else {
        __builtin_amdgcn_s_wait_tensorcnt(0);
      }
    }
    __syncthreads();
    const bf16_t* bb = bsh[c & 1];
    int kbase = c * KC;
#pragma unroll
    for (int kk = 0; kk < KC; kk += 32) {
      v16bf a = load_a_frag(A, K, m0, kbase + kk, lane);
      v16bf b = load_b_frag(bb, KC, n0l, kk, lane);
      acc = wmma_bf16(a, b, acc);
    }
    __syncthreads();
  }

  int col = n0b + n0l + (lane & 15);
  float bn = bias[col];
#pragma unroll
  for (int r = 0; r < 8; ++r) {
    int row = m0 + r + ((lane >> 4) << 3);
    C[(size_t)row * N + col] = acc[r] + bn;
  }
}

// ---------------------------------------------------------------------------
// Encoder LSTM (fwd + bwd): one persistent workgroup per direction.
// H=512, 4H=2048. 8 waves; each wave owns NJ=4 hidden-column tiles and all
// four gate tiles for them -> gate math fully in registers.
// Hidden state double-buffered in LDS as bf16 (WMMA A operand), rows 4..15 = 0.
// ---------------------------------------------------------------------------
__global__ __launch_bounds__(256) void k_lstm_enc(const float* __restrict__ xw_f,
                                                  const float* __restrict__ xw_b,
                                                  const bf16_t* __restrict__ WhT_f,
                                                  const bf16_t* __restrict__ WhT_b,
                                                  float* __restrict__ enc_out) {
  const int Hh = 512, G4 = 2048, T = 128, Gt = 32, NJ = 4;
  __shared__ bf16_t hbuf[2][16 * 512];   // 32 KB
  int dir = blockIdx.x;
  const float*  xw  = dir ? xw_b  : xw_f;
  const bf16_t* WhT = dir ? WhT_b : WhT_f;
  int lane = threadIdx.x & 31, wave = threadIdx.x >> 5;

  for (int i = threadIdx.x; i < 2 * 16 * 512; i += 256)
    ((bf16_t*)hbuf)[i] = (bf16_t)0.0f;
  float c_reg[NJ][4];
#pragma unroll
  for (int jj = 0; jj < NJ; ++jj)
#pragma unroll
    for (int r = 0; r < 4; ++r) c_reg[jj][r] = 0.0f;
  __syncthreads();

  for (int s = 0; s < T; ++s) {
    int t = dir ? (T - 1 - s) : s;   // bwd scans (and emits) reversed time
    const bf16_t* hprev = hbuf[s & 1];
    bf16_t* hnext = hbuf[(s + 1) & 1];
    for (int jj = 0; jj < NJ; ++jj) {
      int j = wave * NJ + jj;
      v8f acc[4];
#pragma unroll
      for (int g = 0; g < 4; ++g)
#pragma unroll
        for (int i = 0; i < 8; ++i) acc[g][i] = 0.0f;
      for (int k0 = 0; k0 < Hh; k0 += 32) {
        v16bf a = load_a_frag(hprev, Hh, 0, k0, lane);
#pragma unroll
        for (int g = 0; g < 4; ++g) {
          v16bf b = load_b_frag(WhT, Hh, (g * Gt + j) * 16, k0, lane);
          acc[g] = wmma_bf16(a, b, acc[g]);
        }
      }
      if (lane < 16) {
        int hcol = j * 16 + lane;
#pragma unroll
        for (int r = 0; r < 4; ++r) {            // r == batch index (M rows 0..3)
          size_t xrow = (size_t)(r * T + t) * G4;
          float zi = acc[0][r] + xw[xrow + 0 * Hh + hcol];
          float zf = acc[1][r] + xw[xrow + 1 * Hh + hcol];
          float zg = acc[2][r] + xw[xrow + 2 * Hh + hcol];
          float zo = acc[3][r] + xw[xrow + 3 * Hh + hcol];
          float c = sigmoidf_(zf) * c_reg[jj][r] + sigmoidf_(zi) * tanhf(zg);
          c_reg[jj][r] = c;
          float h = sigmoidf_(zo) * tanhf(c);
          hnext[r * Hh + hcol] = (bf16_t)h;
          enc_out[(size_t)(r * T + t) * 1024 + dir * Hh + hcol] = h;
        }
      }
    }
    __syncthreads();
  }
}

// ---------------------------------------------------------------------------
// Decoder LSTM: D=1024, 4D=4096, h0 = enc_out[:,T-1,:], c0 = 0.
// ---------------------------------------------------------------------------
__global__ __launch_bounds__(256) void k_lstm_dec(const float* __restrict__ xw,
                                                  const bf16_t* __restrict__ WhT,
                                                  const float* __restrict__ enc_out,
                                                  float* __restrict__ dec_out) {
  const int Dd = 1024, G4 = 4096, T = 128, Gt = 64, NJ = 8;
  __shared__ bf16_t hbuf[2][16 * 1024];  // 64 KB
  int lane = threadIdx.x & 31, wave = threadIdx.x >> 5;

  for (int i = threadIdx.x; i < 2 * 16 * 1024; i += 256)
    ((bf16_t*)hbuf)[i] = (bf16_t)0.0f;
  for (int i = threadIdx.x; i < 4 * 1024; i += 256) {
    int b = i >> 10, d = i & 1023;
    hbuf[0][b * Dd + d] = (bf16_t)enc_out[(size_t)(b * T + (T - 1)) * Dd + d];
  }
  float c_reg[NJ][4];
#pragma unroll
  for (int jj = 0; jj < NJ; ++jj)
#pragma unroll
    for (int r = 0; r < 4; ++r) c_reg[jj][r] = 0.0f;
  __syncthreads();

  for (int t = 0; t < T; ++t) {
    const bf16_t* hprev = hbuf[t & 1];
    bf16_t* hnext = hbuf[(t + 1) & 1];
    for (int jj = 0; jj < NJ; ++jj) {
      int j = wave * NJ + jj;
      v8f acc[4];
#pragma unroll
      for (int g = 0; g < 4; ++g)
#pragma unroll
        for (int i = 0; i < 8; ++i) acc[g][i] = 0.0f;
      for (int k0 = 0; k0 < Dd; k0 += 32) {
        v16bf a = load_a_frag(hprev, Dd, 0, k0, lane);
#pragma unroll
        for (int g = 0; g < 4; ++g) {
          v16bf b = load_b_frag(WhT, Dd, (g * Gt + j) * 16, k0, lane);
          acc[g] = wmma_bf16(a, b, acc[g]);
        }
      }
      if (lane < 16) {
        int hcol = j * 16 + lane;
#pragma unroll
        for (int r = 0; r < 4; ++r) {
          size_t xrow = (size_t)(r * T + t) * G4;
          float zi = acc[0][r] + xw[xrow + 0 * Dd + hcol];
          float zf = acc[1][r] + xw[xrow + 1 * Dd + hcol];
          float zg = acc[2][r] + xw[xrow + 2 * Dd + hcol];
          float zo = acc[3][r] + xw[xrow + 3 * Dd + hcol];
          float c = sigmoidf_(zf) * c_reg[jj][r] + sigmoidf_(zi) * tanhf(zg);
          c_reg[jj][r] = c;
          float h = sigmoidf_(zo) * tanhf(c);
          hnext[r * Dd + hcol] = (bf16_t)h;
          dec_out[(size_t)(r * T + t) * Dd + hcol] = h;
        }
      }
    }
    __syncthreads();
  }
}

// ---------------------------------------------------------------------------
// Bahdanau-style attention. One block per (b,t); output bf16 for final GEMM.
// ---------------------------------------------------------------------------
__global__ __launch_bounds__(256) void k_attn(const float* __restrict__ dec_out,
                                              const float* __restrict__ enc_out,
                                              const float* __restrict__ scale,
                                              bf16_t* __restrict__ attn_bf) {
  const int T = 128, D = 1024;
  int bt = blockIdx.x;
  int b = bt >> 7;
  __shared__ float dvec[1024];
  __shared__ float sc[128];
  int tid = threadIdx.x, lane = tid & 31, wave = tid >> 5;

  for (int d = tid; d < D; d += 256) dvec[d] = dec_out[(size_t)bt * D + d];
  __syncthreads();

  for (int s = wave; s < T; s += 8) {
    const float* er = enc_out + (size_t)(b * T + s) * D;
    float acc = 0.0f;
    for (int d = lane; d < D; d += 32)
      acc += tanhf(dvec[d] + er[d]) * scale[d];
#pragma unroll
    for (int off = 16; off > 0; off >>= 1) acc += __shfl_xor(acc, off, 32);
    if (lane == 0) sc[s] = acc;
  }
  __syncthreads();

  float mx = -1e30f;
  for (int s = 0; s < T; ++s) mx = fmaxf(mx, sc[s]);
  float sum = 0.0f;
  for (int s = 0; s < T; ++s) sum += __expf(sc[s] - mx);
  float inv = 1.0f / sum;
  __syncthreads();
  if (tid < T) sc[tid] = __expf(sc[tid] - mx) * inv;
  __syncthreads();

  for (int d = tid; d < D; d += 256) {
    float a = 0.0f;
    for (int s = 0; s < T; ++s) a += sc[s] * enc_out[(size_t)(b * T + s) * D + d];
    attn_bf[(size_t)bt * D + d] = (bf16_t)a;
  }
}

// ---------------------------------------------------------------------------
// In-place row softmax over N columns (N = 32000), one block per row.
// ---------------------------------------------------------------------------
__global__ __launch_bounds__(256) void k_softmax_rows(float* __restrict__ x, int N) {
  float* p = x + (size_t)blockIdx.x * N;
  __shared__ float redm[8];
  __shared__ float reds[8];
  int tid = threadIdx.x, lane = tid & 31, wave = tid >> 5;

  float mx = -1e30f;
  for (int i = tid; i < N; i += 256) mx = fmaxf(mx, p[i]);
#pragma unroll
  for (int off = 16; off > 0; off >>= 1) mx = fmaxf(mx, __shfl_xor(mx, off, 32));
  if (lane == 0) redm[wave] = mx;
  __syncthreads();
  mx = redm[0];
#pragma unroll
  for (int w = 1; w < 8; ++w) mx = fmaxf(mx, redm[w]);

  float sum = 0.0f;
  for (int i = tid; i < N; i += 256) { float e = __expf(p[i] - mx); p[i] = e; sum += e; }
#pragma unroll
  for (int off = 16; off > 0; off >>= 1) sum += __shfl_xor(sum, off, 32);
  if (lane == 0) reds[wave] = sum;
  __syncthreads();
  sum = 0.0f;
#pragma unroll
  for (int w = 0; w < 8; ++w) sum += reds[w];
  float inv = 1.0f / sum;
  for (int i = tid; i < N; i += 256) p[i] *= inv;
}

// ---------------------------------------------------------------------------
extern "C" void kernel_launch(void* const* d_in, const int* in_sizes, int n_in,
                              void* d_out, int out_size, void* d_ws, size_t ws_size,
                              hipStream_t stream) {
  (void)in_sizes; (void)n_in; (void)out_size; (void)ws_size;
  const int*   input_seq  = (const int*)  d_in[0];
  const int*   output_seq = (const int*)  d_in[1];
  const float* enc_emb    = (const float*)d_in[2];
  const float* dec_emb    = (const float*)d_in[3];
  const float* Wx_f = (const float*)d_in[4];
  const float* Wh_f = (const float*)d_in[5];
  const float* b_f  = (const float*)d_in[6];
  const float* Wx_b = (const float*)d_in[7];
  const float* Wh_b = (const float*)d_in[8];
  const float* b_b  = (const float*)d_in[9];
  const float* Wx_d = (const float*)d_in[10];
  const float* Wh_d = (const float*)d_in[11];
  const float* b_d  = (const float*)d_in[12];
  const float* attn_scale = (const float*)d_in[13];
  const float* Wo = (const float*)d_in[14];
  const float* bo = (const float*)d_in[15];
  float* out = (float*)d_out;

  const int B = 4, T = 128, E = 512, H = 512, D = 1024, V = 32000;
  const int S = B * T;   // 512 rows

  char* ws = (char*)d_ws;
  size_t off = 0;
  auto alloc = [&](size_t bytes) -> void* {
    void* p = ws + off;
    off = (off + bytes + 255) & ~(size_t)255;
    return p;
  };
  bf16_t* WxTf  = (bf16_t*)alloc((size_t)4*H * E * 2);      // [2048,512]
  bf16_t* WhTf  = (bf16_t*)alloc((size_t)4*H * H * 2);
  bf16_t* WxTb  = (bf16_t*)alloc((size_t)4*H * E * 2);
  bf16_t* WhTb  = (bf16_t*)alloc((size_t)4*H * H * 2);
  bf16_t* WxTd  = (bf16_t*)alloc((size_t)4*D * E * 2);      // [4096,512]
  bf16_t* WhTd  = (bf16_t*)alloc((size_t)4*D * D * 2);      // [4096,1024]
  bf16_t* WoT   = (bf16_t*)alloc((size_t)V * D * 2);        // [32000,1024]
  bf16_t* enc_x = (bf16_t*)alloc((size_t)S * E * 2);
  bf16_t* dec_x = (bf16_t*)alloc((size_t)S * E * 2);
  float*  xw_f  = (float*) alloc((size_t)S * 4*H * 4);
  float*  xw_b  = (float*) alloc((size_t)S * 4*H * 4);
  float*  xw_d  = (float*) alloc((size_t)S * 4*D * 4);
  float*  enc_out = (float*)alloc((size_t)S * D * 4);       // [B,T,1024]
  float*  dec_out = (float*)alloc((size_t)S * D * 4);
  bf16_t* attn_bf = (bf16_t*)alloc((size_t)S * D * 2);

  dim3 blk(256);
  // --- weight transposes (f32 -> bf16, [rows,cols] -> [cols,rows]) ---
  k_transpose_to_bf16<<<dim3(4*H/32, E/32), blk, 0, stream>>>(Wx_f, WxTf, E, 4*H);
  k_transpose_to_bf16<<<dim3(4*H/32, H/32), blk, 0, stream>>>(Wh_f, WhTf, H, 4*H);
  k_transpose_to_bf16<<<dim3(4*H/32, E/32), blk, 0, stream>>>(Wx_b, WxTb, E, 4*H);
  k_transpose_to_bf16<<<dim3(4*H/32, H/32), blk, 0, stream>>>(Wh_b, WhTb, H, 4*H);
  k_transpose_to_bf16<<<dim3(4*D/32, E/32), blk, 0, stream>>>(Wx_d, WxTd, E, 4*D);
  k_transpose_to_bf16<<<dim3(4*D/32, D/32), blk, 0, stream>>>(Wh_d, WhTd, D, 4*D);
  k_transpose_to_bf16<<<dim3(V/32,   D/32), blk, 0, stream>>>(Wo,   WoT,  D, V);

  // --- embeddings ---
  k_embed_bf16<<<dim3(S), blk, 0, stream>>>(input_seq,  enc_emb, enc_x, E);
  k_embed_bf16<<<dim3(S), blk, 0, stream>>>(output_seq, dec_emb, dec_x, E);

  // --- input projections xw = x @ Wx + b (WMMA, TDM-staged B) ---
  k_gemm_bf16_tdm<<<dim3(4*H/64, S/32), blk, 0, stream>>>(enc_x, WxTf, b_f, xw_f, S, 4*H, E);
  k_gemm_bf16_tdm<<<dim3(4*H/64, S/32), blk, 0, stream>>>(enc_x, WxTb, b_b, xw_b, S, 4*H, E);
  k_gemm_bf16_tdm<<<dim3(4*D/64, S/32), blk, 0, stream>>>(dec_x, WxTd, b_d, xw_d, S, 4*D, E);

  // --- recurrences ---
  k_lstm_enc<<<dim3(2), blk, 0, stream>>>(xw_f, xw_b, WhTf, WhTb, enc_out);
  k_lstm_dec<<<dim3(1), blk, 0, stream>>>(xw_d, WhTd, enc_out, dec_out);

  // --- attention ---
  k_attn<<<dim3(S), blk, 0, stream>>>(dec_out, enc_out, attn_scale, attn_bf);

  // --- output projection (dominant GEMM, TDM-staged B) into d_out, softmax ---
  k_gemm_bf16_tdm<<<dim3(V/64, S/32), blk, 0, stream>>>(attn_bf, WoT, bo, out, S, V, D);
  k_softmax_rows<<<dim3(S), blk, 0, stream>>>(out, V);
}
